// Conv2DFastKAN_19284403159602
// MI455X (gfx1250) — matrix operand order repaired and checked by reading, not compiled, not run on hardware
//
#include <hip/hip_runtime.h>
#include <hip/hip_bf16.h>

typedef __attribute__((ext_vector_type(16))) _Float16 v16h;
typedef __attribute__((ext_vector_type(8)))  float    v8f;

#define Bsz 8
#define Cin 16
#define Himg 64
#define Wimg 64
#define Oc 32
#define Dfeat 144            // C*K*K
#define Gg 8
#define DG 1152              // D*G
#define NKC (DG / 32)        // 36 K-chunks
#define NPATCH 32768         // B*64*64
#define L2E 1.4426950408889634f
// denom = 4/7, h = denom^2
#define INV_H_L2E (L2E * 49.0f / 16.0f)       // L2E / h
#define R_COEF    (3.5f * L2E)                // (2/denom) * L2E

// ---------------- kernel 1: Wsum[o][k] = sum_d W[o][d][k], fp32 -> fp16 ------
__global__ __launch_bounds__(256) void wsum_kernel(const float* __restrict__ W,
                                                   _Float16* __restrict__ wsum) {
    int t = blockIdx.x * 256 + threadIdx.x;          // 36864 threads
    if (t >= Oc * DG) return;
    int o = t / DG, k = t - o * DG;
    const float* wp = W + (size_t)o * Dfeat * DG + k;
    float s = 0.f;
    #pragma unroll 4
    for (int d = 0; d < Dfeat; ++d) s += wp[(size_t)d * DG];
    wsum[t] = (_Float16)s;
}

// ---------------- kernel 2: fused unfold + RBF + GEMM (WMMA) -----------------
// 1 workgroup = 1 image row (64 patches). 8 waves: 4 M-tiles x 2 N-tiles.
__global__ __launch_bounds__(256) void fastkan_gemm(const float* __restrict__ x,
                                                    const _Float16* __restrict__ wsum,
                                                    float* __restrict__ out,
                                                    float* __restrict__ partial) {
    __shared__ float xs[Cin * 3 * 66];   // [c][dr][col+1], zero-padded cols/rows
    __shared__ float shs[8][32];
    __shared__ float shq[8][32];

    const int wg  = blockIdx.x;          // 0..511
    const int b   = wg >> 6;
    const int row = wg & 63;
    const int tid = threadIdx.x;

    // pull Wsum (72KB, shared by all WGs) toward this WGP: global_prefetch_b8
    {
        const char* wp = (const char*)wsum;
        for (int i = tid; i < 576; i += 256)
            __builtin_prefetch(wp + i * 128, 0, 3);
    }

    // stage input slab: rows row-1..row+1, all 16 channels, padded cols
    for (int i = tid; i < Cin * 3 * 66; i += 256) {
        int c  = i / (3 * 66);
        int rr = (i % (3 * 66)) / 66;
        int cc = (i % 66) - 1;           // image col
        int ir = row + rr - 1;
        float v = 0.f;
        if (ir >= 0 && ir < Himg && cc >= 0 && cc < Wimg)
            v = x[(((size_t)b * Cin + c) * Himg + ir) * Wimg + cc];
        xs[i] = v;
    }
    __syncthreads();

    const int wave = tid >> 5;
    const int lane = tid & 31;
    const int mt   = wave >> 1;          // M tile (patch cols)
    const int nt   = wave & 1;           // N tile (out channels)
    const int colm = mt * 16 + (lane & 15);        // A-matrix row M -> image col
    const int dsel = (lane < 16) ? 0 : 1;          // K layout: lanes 0-15 own d_rel {0,1}
    const int o    = nt * 16 + (lane & 15);        // B-matrix col N -> out channel
    const _Float16* brow = wsum + (size_t)o * DG + ((lane < 16) ? 0 : 16);

    // s_g = exp(-grid_g^2/h), grid_g = -2 + g*4/7   (8 regs, computed once)
    float sg[Gg];
    #pragma unroll
    for (int g = 0; g < Gg; ++g) {
        float gv = -2.0f + (float)g * (4.0f / 7.0f);
        sg[g] = __builtin_amdgcn_exp2f(-gv * gv * INV_H_L2E);
    }

    v8f acc = {};
    // software-pipelined B fetch: load chunk kc+1 before issuing WMMA for kc
    v16h bm = *(const v16h*)(brow);
    #pragma unroll 2
    for (int kc = 0; kc < NKC; ++kc) {
        int kn = (kc + 1 < NKC) ? (kc + 1) : 0;    // clamped (value unused past end)
        v16h bnext = *(const v16h*)(brow + kn * 32);

        // Build A (16x32 f16) directly in WMMA register layout.
        // Per lane: two feature values d, 8 RBF grids each.
        v16h a;
        #pragma unroll
        for (int hfl = 0; hfl < 2; ++hfl) {
            int d_abs = kc * 4 + 2 * dsel + hfl;   // absolute feature index
            int c   = d_abs / 9;
            int rem = d_abs - c * 9;
            int kh  = rem / 3;
            int kw  = rem - kh * 3;
            float p = xs[(c * 3 + kh) * 66 + colm + kw];   // padded slab: no branch
            // basis_g = exp(-p^2/h) * r^g * s_g,  r = exp(2p/denom)
            float a0 = __builtin_amdgcn_exp2f(-p * p * INV_H_L2E);
            float rr = __builtin_amdgcn_exp2f(p * R_COEF);
            float v  = a0;
            #pragma unroll
            for (int g = 0; g < Gg; ++g) {
                a[hfl * 8 + g] = (_Float16)(v * sg[g]);
                v *= rr;
            }
        }
        acc = __builtin_amdgcn_wmma_f32_16x16x32_f16(false, a, false, bm,
                                                     (short)0, acc, false, false);
        bm = bnext;
    }

    // C layout: VGPR r, lane L -> M = r + (L<16?0:8), N = L&15 (channel fixed/lane)
    float s = 0.f, q = 0.f;
    int mbase = mt * 16 + ((lane < 16) ? 0 : 8);
    #pragma unroll
    for (int r = 0; r < 8; ++r) {
        float val = acc[r];
        s += val; q += val * val;
        int m = mbase + r;              // image col of this patch
        out[(((size_t)b * Oc + o) << 12) + (row << 6) + m] = val;
    }
    shs[wave][lane] = s;
    shq[wave][lane] = q;
    __syncthreads();

    // deterministic per-channel partials for this workgroup
    if (tid < 32) {
        int nt2 = tid >> 4, nl = tid & 15;
        float S = 0.f, Q = 0.f;
        #pragma unroll
        for (int mt2 = 0; mt2 < 4; ++mt2) {
            int w = mt2 * 2 + nt2;
            S += shs[w][nl] + shs[w][nl + 16];
            Q += shq[w][nl] + shq[w][nl + 16];
        }
        partial[wg * 64 + tid]      = S;
        partial[wg * 64 + 32 + tid] = Q;
    }
}

// ---------------- kernel 3: reduce partials -> per-channel scale/shift -------
__global__ __launch_bounds__(256) void bn_reduce(const float* __restrict__ partial,
                                                 const float* __restrict__ gamma,
                                                 const float* __restrict__ beta,
                                                 float* __restrict__ coef) {
    __shared__ float sh[512];
    int o = blockIdx.x, t = threadIdx.x;
    float s = partial[t * 64 + o]      + partial[(t + 256) * 64 + o];
    float q = partial[t * 64 + 32 + o] + partial[(t + 256) * 64 + 32 + o];
    sh[t] = s; sh[256 + t] = q;
    __syncthreads();
    for (int st = 128; st > 0; st >>= 1) {
        if (t < st) { sh[t] += sh[t + st]; sh[256 + t] += sh[256 + t + st]; }
        __syncthreads();
    }
    if (t == 0) {
        float mean = sh[0]   * (1.0f / (float)NPATCH);
        float var  = sh[256] * (1.0f / (float)NPATCH) - mean * mean;
        float a = gamma[o] * rsqrtf(var + 1e-5f);
        coef[2 * o]     = a;
        coef[2 * o + 1] = beta[o] - mean * a;
    }
}

// ---------------- kernel 4: in-place BN apply --------------------------------
__global__ __launch_bounds__(256) void bn_apply(float* __restrict__ out,
                                                const float* __restrict__ coef,
                                                int n) {
    int idx = blockIdx.x * 256 + threadIdx.x;
    if (idx >= n) return;
    int o = (idx >> 12) & 31;
    out[idx] = out[idx] * coef[2 * o] + coef[2 * o + 1];
}

extern "C" void kernel_launch(void* const* d_in, const int* in_sizes, int n_in,
                              void* d_out, int out_size, void* d_ws, size_t ws_size,
                              hipStream_t stream) {
    const float* x     = (const float*)d_in[0];
    const float* W     = (const float*)d_in[1];
    const float* gamma = (const float*)d_in[2];
    const float* beta  = (const float*)d_in[3];
    float* out = (float*)d_out;

    char* ws = (char*)d_ws;
    _Float16* wsum  = (_Float16*)(ws);               // 32*1152*2   = 73,728 B
    float* partial  = (float*)(ws + 73728);          // 512*64*4    = 131,072 B
    float* coef     = (float*)(ws + 73728 + 131072); // 64*4        = 256 B

    wsum_kernel <<<(Oc * DG + 255) / 256, 256, 0, stream>>>(W, wsum);
    fastkan_gemm<<<NPATCH / 64, 256, 0, stream>>>(x, wsum, out, partial);
    bn_reduce   <<<Oc, 256, 0, stream>>>(partial, gamma, beta, coef);
    bn_apply    <<<(out_size + 255) / 256, 256, 0, stream>>>(out, coef, out_size);
}
